// win_and_group_tf_13675175870644
// MI455X (gfx1250) — matrix operand order repaired and checked by reading, not compiled, not run on hardware
//
#include <hip/hip_runtime.h>
#include <hip/hip_bf16.h>
#include <math.h>

#define D 128
#define G 64
#define H 8
#define HD 16
#define NGRP 2048
#define MG (NGRP * G)
#define NPTS 100000
#define QSCALE 0.25f
#define LNEPS 1e-5f

typedef __attribute__((ext_vector_type(16))) _Float16 v16h;
typedef __attribute__((ext_vector_type(8)))  float    v8f;

__device__ __forceinline__ v8f wmma32f16(v16h a, v16h b, v8f c) {
  // D = A(16x32 f16) * B(32x16 f16) + C(16x16 f32)
  return __builtin_amdgcn_wmma_f32_16x16x32_f16(false, a, false, b, (short)0, c,
                                                false, false);
}

__device__ __forceinline__ v16h zero16h() {
  v16h z;
#pragma unroll
  for (int i = 0; i < 16; ++i) z[i] = (_Float16)0.f;
  return z;
}

__device__ __forceinline__ v8f zero8f() {
  v8f z = {0.f, 0.f, 0.f, 0.f, 0.f, 0.f, 0.f, 0.f};
  return z;
}

// Async 16-byte global -> LDS copy (GLOBAL_LOAD_ASYNC_TO_LDS_B128, ASYNCcnt).
// VDST = VGPR holding LDS byte address, VADDR = 64-bit global address.
__device__ __forceinline__ void async_copy_b128(const void* gsrc, void* ldst) {
  const unsigned lds = (unsigned)(size_t)ldst;  // low 32 bits of generic ptr = LDS offset
  const unsigned long long g = (unsigned long long)(size_t)gsrc;
  asm volatile("global_load_async_to_lds_b128 %0, %1, off"
               :: "v"(lds), "v"(g) : "memory");
}
__device__ __forceinline__ void async_wait0() {
  asm volatile("s_wait_asynccnt 0x0" ::: "memory");
}

// ---------------------------------------------------------------- utilities

__global__ void k_zero(float* __restrict__ p, long n) {
  long i = (long)blockIdx.x * blockDim.x + threadIdx.x;
  const long stride = (long)gridDim.x * blockDim.x;
  for (; i < n; i += stride) p[i] = 0.f;
}

__global__ void k_copy(const float* __restrict__ a, float* __restrict__ o, long n) {
  long i = (long)blockIdx.x * blockDim.x + threadIdx.x;
  const long stride = (long)gridDim.x * blockDim.x;
  for (; i < n; i += stride) o[i] = a[i];
}

__global__ void k_f32_to_f16(const float* __restrict__ a, _Float16* __restrict__ o, long n) {
  long i = (long)blockIdx.x * blockDim.x + threadIdx.x;
  const long stride = (long)gridDim.x * blockDim.x;
  for (; i < n; i += stride) o[i] = (_Float16)a[i];
}

// Xg[slots[j]] = x[perm[j]]; occ[slots[j]] = 1
__global__ void k_group_scatter(const float* __restrict__ x, const int* __restrict__ perm,
                                const int* __restrict__ slots, float* __restrict__ Xg,
                                int* __restrict__ occ, int n) {
  const long total = (long)n * D;
  const long stride = (long)gridDim.x * blockDim.x;
  for (long e = (long)blockIdx.x * blockDim.x + threadIdx.x; e < total; e += stride) {
    const int j = (int)(e >> 7), c = (int)(e & (D - 1));
    const int s = slots[j], p = perm[j];
    Xg[(long)s * D + c] = x[(long)p * D + c];
    if (c == 0) occ[s] = 1;
  }
}

// dst[perm[j]] = src[slots[j]]
__global__ void k_scatter_set(const int* __restrict__ perm, const int* __restrict__ slots,
                              const float* __restrict__ src, float* __restrict__ dst, int n) {
  const long total = (long)n * D;
  const long stride = (long)gridDim.x * blockDim.x;
  for (long e = (long)blockIdx.x * blockDim.x + threadIdx.x; e < total; e += stride) {
    const int j = (int)(e >> 7), c = (int)(e & (D - 1));
    dst[(long)perm[j] * D + c] = src[(long)slots[j] * D + c];
  }
}

// dst[perm[j]] += src[slots[j]]
__global__ void k_scatter_add(const int* __restrict__ perm, const int* __restrict__ slots,
                              const float* __restrict__ src, float* __restrict__ dst, int n) {
  const long total = (long)n * D;
  const long stride = (long)gridDim.x * blockDim.x;
  for (long e = (long)blockIdx.x * blockDim.x + threadIdx.x; e < total; e += stride) {
    const int j = (int)(e >> 7), c = (int)(e & (D - 1));
    dst[(long)perm[j] * D + c] += src[(long)slots[j] * D + c];
  }
}

// Pack a KxN f32 matrix into WMMA B-fragment order (f16):
// tile (kt, nt) of 32x16; element (kk, nn) -> lane = nn + 16*(kk>>4), idx = kk&15.
__global__ void k_pack_b(const float* __restrict__ W, _Float16* __restrict__ P,
                         int K, int N) {
  const long total = (long)K * N;
  const int ntiles = N >> 4;
  const long stride = (long)gridDim.x * blockDim.x;
  for (long e = (long)blockIdx.x * blockDim.x + threadIdx.x; e < total; e += stride) {
    const int k = (int)(e / N), n = (int)(e % N);
    const int kt = k >> 5, kk = k & 31, nt = n >> 4, nn = n & 15;
    const int lane = nn + ((kk >> 4) << 4);
    const int idx = kk & 15;
    P[((long)(kt * ntiles + nt) * 32 + lane) * 16 + idx] = (_Float16)W[e];
  }
}

// ---------------------------------------------------------------- GEMM (K=128)
// C[M x N] = A[M x 128] * B + bias ; block = 256 thr (8 waves: 4 M x 2 N-pairs)
// Each wave computes TWO 16x16 tiles (reuses each A fragment across 2 WMMAs).
// grid = (M/64, N/64). A f32 global -> f16 LDS staging; B pre-packed frags.
__global__ void k_gemm128(const float* __restrict__ A, const _Float16* __restrict__ Bp,
                          const float* __restrict__ bias, float* __restrict__ Cf,
                          _Float16* __restrict__ Ch, int N, int relu) {
  __shared__ __align__(16) _Float16 sA[64 * 128];  // 16 KB
  const int tid = threadIdx.x;
  const int wave = tid >> 5, lane = tid & 31;
  const long mBlock = (long)blockIdx.x * 64;

  // stage A tile (64 x 128) as f16
  for (int e = tid; e < 64 * 32; e += 256) {
    const int r = e >> 5;
    const int c = (e & 31) << 2;
    const float4 f = *(const float4*)(A + (mBlock + r) * D + c);
    sA[r * D + c + 0] = (_Float16)f.x;
    sA[r * D + c + 1] = (_Float16)f.y;
    sA[r * D + c + 2] = (_Float16)f.z;
    sA[r * D + c + 3] = (_Float16)f.w;
  }
  __syncthreads();

  const int mW = wave >> 1;                              // 0..3
  const int ntBase = blockIdx.y * 4 + (wave & 1) * 2;    // first of 2 n-tiles
  const int ntiles = N >> 4;
  const int mRow = lane & 15, kSel = lane >> 4;
  v8f acc[2];
  acc[0] = zero8f();
  acc[1] = zero8f();
#pragma unroll
  for (int kt = 0; kt < 4; ++kt) {
    v16h a;
    const _Float16* pa = &sA[(mW * 16 + mRow) * D + kt * 32 + kSel * 16];
#pragma unroll
    for (int i = 0; i < 8; ++i) { a[i] = pa[i]; a[8 + i] = pa[8 + i]; }
    v16h b0, b1;
    const _Float16* pb0 = Bp + ((long)(kt * ntiles + ntBase) * 32 + lane) * 16;
    const _Float16* pb1 = Bp + ((long)(kt * ntiles + ntBase + 1) * 32 + lane) * 16;
#pragma unroll
    for (int i = 0; i < 16; ++i) { b0[i] = pb0[i]; b1[i] = pb1[i]; }
    acc[0] = wmma32f16(a, b0, acc[0]);
    acc[1] = wmma32f16(a, b1, acc[1]);
  }
  const int nn = lane & 15, mOff = (lane >> 4) * 8;
#pragma unroll
  for (int t = 0; t < 2; ++t) {
    const int col = (ntBase + t) * 16 + nn;
    const float bv = bias ? bias[col] : 0.f;
#pragma unroll
    for (int r = 0; r < 8; ++r) {
      float v = acc[t][r] + bv;
      if (relu) v = fmaxf(v, 0.f);
      const long row = mBlock + mW * 16 + mOff + r;
      if (Cf) Cf[row * N + col] = v;
      if (Ch) Ch[row * N + col] = (_Float16)v;
    }
  }
}

// ---------------------------------------------------------------- attention
// grid (NGRP, H), block 128 (4 waves). q/k/v head slices async-staged to LDS,
// scores(64x64, K=16 padded to 32) -> softmax (query-row mask per reference
// broadcast semantics) -> probs@v.
__global__ void k_attn(const _Float16* __restrict__ q16, const _Float16* __restrict__ kv16,
                       const int* __restrict__ occ, float* __restrict__ feat) {
  __shared__ __align__(16) _Float16 sq[G * HD], sk[G * HD], sv[G * HD];  // 3 * 2 KB
  __shared__ float    sS[G * G];                                        // 16 KB
  __shared__ _Float16 sP[G * G];                                        // 8 KB
  const int g = blockIdx.x, h = blockIdx.y;
  const int tid = threadIdx.x, wave = tid >> 5, lane = tid & 31;

  {  // async stage: 64 rows x 2 b128 chunks per matrix, 128 threads -> 1 each
    const int r = tid >> 1, chunk = tid & 1;
    const long row = (long)g * G + r;
    const int co = chunk * 8;  // halves
    async_copy_b128(q16 + row * D + h * HD + co,            &sq[r * HD + co]);
    async_copy_b128(kv16 + row * (2 * D) + h * HD + co,     &sk[r * HD + co]);
    async_copy_b128(kv16 + row * (2 * D) + D + h * HD + co, &sv[r * HD + co]);
  }
  async_wait0();
  __syncthreads();

  const int mRow = lane & 15, kSel = lane >> 4;
  const int nn = lane & 15, mOff = (lane >> 4) * 8;

  {  // scores: wave handles m-tile = wave; QSCALE folded into score write
    v16h a = zero16h();  // K 16..31 stay zero (hd=16 padded)
    const _Float16* pa = &sq[(wave * 16 + mRow) * HD];
#pragma unroll
    for (int i = 0; i < 16; ++i) a[i] = kSel ? (_Float16)0.f : pa[i];
#pragma unroll
    for (int ntt = 0; ntt < 4; ++ntt) {
      v16h b = zero16h();  // lanes 16-31 carry K>=16 -> zero
      if (lane < 16) {
        const _Float16* pb = &sk[(ntt * 16 + nn) * HD];
#pragma unroll
        for (int i = 0; i < 16; ++i) b[i] = pb[i];
      }
      v8f s = wmma32f16(a, b, zero8f());
#pragma unroll
      for (int r = 0; r < 8; ++r)
        sS[(wave * 16 + mOff + r) * G + ntt * 16 + nn] = s[r] * QSCALE;
    }
  }
  __syncthreads();

  if (tid < G) {  // softmax per query row
    const int occR = occ[g * G + tid];
    float mx = -1e30f;
    for (int j = 0; j < G; ++j) {
      const float s = occR ? sS[tid * G + j] : -1e9f;
      mx = fmaxf(mx, s);
    }
    float sum = 0.f;
    for (int j = 0; j < G; ++j) {
      const float s = occR ? sS[tid * G + j] : -1e9f;
      sum += __expf(s - mx);
    }
    const float inv = 1.f / sum;
    for (int j = 0; j < G; ++j) {
      const float s = occR ? sS[tid * G + j] : -1e9f;
      sP[tid * G + j] = (_Float16)(__expf(s - mx) * inv);
    }
  }
  __syncthreads();

  v8f acc = zero8f();
#pragma unroll
  for (int kt = 0; kt < 2; ++kt) {
    v16h a;
    const _Float16* pa = &sP[(wave * 16 + mRow) * G + kt * 32 + kSel * 16];
#pragma unroll
    for (int i = 0; i < 8; ++i) { a[i] = pa[i]; a[8 + i] = pa[8 + i]; }
    v16h b;
    const int kHalf = (lane >> 4) << 4;
#pragma unroll
    for (int i = 0; i < 16; ++i) b[i] = sv[(kt * 32 + kHalf + i) * HD + nn];
    acc = wmma32f16(a, b, acc);
  }
#pragma unroll
  for (int r = 0; r < 8; ++r)
    feat[((long)g * G + wave * 16 + mOff + r) * D + h * HD + nn] = acc[r];
}

// ---------------------------------------------------------------- layernorm
// out = LN(X + R) * g + b ; one wave (32 lanes) per row of 128.
__global__ void k_ln_res(const float* __restrict__ X, const float* __restrict__ R,
                         const float* __restrict__ gg, const float* __restrict__ bb,
                         float* __restrict__ out, long rows) {
  const long row = (long)blockIdx.x * 8 + (threadIdx.x >> 5);
  if (row >= rows) return;
  const int lane = threadIdx.x & 31;
  float v[4];
  float s = 0.f;
#pragma unroll
  for (int i = 0; i < 4; ++i) {
    const long idx = row * D + lane * 4 + i;
    v[i] = X[idx] + (R ? R[idx] : 0.f);
    s += v[i];
  }
#pragma unroll
  for (int o = 16; o > 0; o >>= 1) s += __shfl_xor(s, o, 32);
  const float m = s * (1.f / D);
  float q = 0.f;
#pragma unroll
  for (int i = 0; i < 4; ++i) { const float d = v[i] - m; q += d * d; }
#pragma unroll
  for (int o = 16; o > 0; o >>= 1) q += __shfl_xor(q, o, 32);
  const float inv = rsqrtf(q * (1.f / D) + LNEPS);
#pragma unroll
  for (int i = 0; i < 4; ++i) {
    const int c = lane * 4 + i;
    out[row * D + c] = (v[i] - m) * inv * gg[c] + bb[c];
  }
}

// ---------------------------------------------------------------- conv (gathered GEMM)
// out[n,:] = gelu(LN(sum_k pad[nbr[n,k]] @ W[k])) ; K=27*128=3456, N=128.
// A rows gathered from f16 pad (row NPTS = zeros) straight into LDS via
// async global->LDS b128 copies; block 256 (8 waves = 8 N-tiles), 16 pts/block.
__global__ void k_conv(const _Float16* __restrict__ pad16, const int* __restrict__ nbr,
                       const _Float16* __restrict__ Wp, const float* __restrict__ lng,
                       const float* __restrict__ lnb, float* __restrict__ out) {
  __shared__ __align__(16) _Float16 sA[16 * 128];  // 4 KB
  __shared__ float sO[16 * 128];                   // 8 KB
  const int tid = threadIdx.x, wave = tid >> 5, lane = tid & 31;
  const int mBase = blockIdx.x * 16;
  const int mRow = lane & 15, kSel = lane >> 4;
  const int nn = lane & 15, mOff = (lane >> 4) * 8;
  const int gr = tid >> 4, gchunk = tid & 15;  // gather row / 16B chunk
  const int gi = mBase + gr;
  v8f acc = zero8f();

  for (int k = 0; k < 27; ++k) {
    __syncthreads();  // all waves done reading sA from previous iteration
    {
      int src = NPTS;  // zero pad row
      if (gi < NPTS) {
        const int t = nbr[(long)gi * 27 + k];
        src = (t < 0) ? NPTS : t;
      }
      async_copy_b128(pad16 + (long)src * D + gchunk * 8, &sA[gr * D + gchunk * 8]);
    }
    async_wait0();
    __syncthreads();
#pragma unroll
    for (int kt = 0; kt < 4; ++kt) {
      v16h a;
      const _Float16* pa = &sA[mRow * D + kt * 32 + kSel * 16];
#pragma unroll
      for (int i = 0; i < 8; ++i) { a[i] = pa[i]; a[8 + i] = pa[8 + i]; }
      v16h b;
      const _Float16* pb = Wp + ((long)((k * 4 + kt) * 8 + wave) * 32 + lane) * 16;
#pragma unroll
      for (int i = 0; i < 16; ++i) b[i] = pb[i];
      acc = wmma32f16(a, b, acc);
    }
  }
  __syncthreads();
#pragma unroll
  for (int r = 0; r < 8; ++r) sO[(mOff + r) * D + wave * 16 + nn] = acc[r];
  __syncthreads();

  for (int rr = wave; rr < 16; rr += 8) {  // fused LN + exact GELU
    float v[4];
    float s = 0.f;
#pragma unroll
    for (int i = 0; i < 4; ++i) { v[i] = sO[rr * D + lane * 4 + i]; s += v[i]; }
#pragma unroll
    for (int o = 16; o > 0; o >>= 1) s += __shfl_xor(s, o, 32);
    const float m = s * (1.f / D);
    float q = 0.f;
#pragma unroll
    for (int i = 0; i < 4; ++i) { const float d = v[i] - m; q += d * d; }
#pragma unroll
    for (int o = 16; o > 0; o >>= 1) q += __shfl_xor(q, o, 32);
    const float inv = rsqrtf(q * (1.f / D) + LNEPS);
    const int row = mBase + rr;
#pragma unroll
    for (int i = 0; i < 4; ++i) {
      const int c = lane * 4 + i;
      const float xn = (v[i] - m) * inv * lng[c] + lnb[c];
      const float ge = 0.5f * xn * (1.f + erff(xn * 0.70710678118654752f));
      if (row < NPTS) out[(long)row * D + c] = ge;
    }
  }
}

// ---------------------------------------------------------------- launch

extern "C" void kernel_launch(void* const* d_in, const int* in_sizes, int n_in,
                              void* d_out, int out_size, void* d_ws, size_t ws_size,
                              hipStream_t stream) {
  (void)in_sizes; (void)n_in; (void)out_size; (void)ws_size;
  const float* x    = (const float*)d_in[0];
  const float* Wq   = (const float*)d_in[1];
  const float* bq   = (const float*)d_in[2];
  const float* Wkv  = (const float*)d_in[3];
  const float* bkv  = (const float*)d_in[4];
  const float* ln1g = (const float*)d_in[5];
  const float* ln1b = (const float*)d_in[6];
  const float* W1   = (const float*)d_in[7];
  const float* b1   = (const float*)d_in[8];
  const float* W2   = (const float*)d_in[9];
  const float* b2   = (const float*)d_in[10];
  const float* ln2g = (const float*)d_in[11];
  const float* ln2b = (const float*)d_in[12];
  const float* convW= (const float*)d_in[13];
  const float* clng = (const float*)d_in[14];
  const float* clnb = (const float*)d_in[15];
  const int* permX  = (const int*)d_in[16];
  const int* slotsX = (const int*)d_in[17];
  const int* permY  = (const int*)d_in[18];
  const int* slotsY = (const int*)d_in[19];
  const int* nbr    = (const int*)d_in[20];

  char* ws = (char*)d_ws;
  size_t off = 0;
  auto take = [&](size_t bytes) -> char* {
    char* p = ws + off;
    off += (bytes + 255) & ~(size_t)255;
    return p;
  };
  const size_t actB = (size_t)MG * D * sizeof(float);
  float* buf0 = (float*)take(actB);   // Xg -> Xo
  float* buf1 = (float*)take(actB);   // Yg -> Yo
  float* buf2 = (float*)take(actB);   // X1
  float* buf3 = (float*)take(actB);   // Y1
  float* featb = (float*)take(actB);  // attn feat / x2
  float* xln  = (float*)take(actB);   // LN1 output
  float* hbuf = (float*)take(actB);   // FFN hidden
  _Float16* q16  = (_Float16*)take((size_t)MG * D * 2);
  _Float16* kv16 = (_Float16*)take((size_t)MG * 2 * D * 2);
  int* occX = (int*)take((size_t)MG * 4);
  int* occY = (int*)take((size_t)MG * 4);
  float* xTF = (float*)take((size_t)NPTS * D * sizeof(float));
  _Float16* pad16 = (_Float16*)take((size_t)(NPTS + 1) * D * 2);  // f16 + zero row
  _Float16* WqP  = (_Float16*)take((size_t)4 * D * D * 2);
  _Float16* WkvP = (_Float16*)take((size_t)4 * D * 2 * D * 2);
  _Float16* W1P  = (_Float16*)take((size_t)4 * D * D * 2);
  _Float16* W2P  = (_Float16*)take((size_t)4 * D * D * 2);
  _Float16* convP= (_Float16*)take((size_t)27 * D * D * 2);

  // group + occupancy
  k_zero<<<4096, 256, 0, stream>>>(buf0, (long)MG * D);
  k_zero<<<4096, 256, 0, stream>>>(buf1, (long)MG * D);
  k_zero<<<256, 256, 0, stream>>>((float*)occX, (long)MG);
  k_zero<<<256, 256, 0, stream>>>((float*)occY, (long)MG);
  k_group_scatter<<<8192, 256, 0, stream>>>(x, permX, slotsX, buf0, occX, NPTS);
  k_group_scatter<<<8192, 256, 0, stream>>>(x, permY, slotsY, buf1, occY, NPTS);

  // pack weights to WMMA B-fragment layout (f16)
  for (int i = 0; i < 4; ++i) {
    k_pack_b<<<64, 256, 0, stream>>>(Wq  + (size_t)i * D * D,      WqP  + (size_t)i * D * D,     D, D);
    k_pack_b<<<128, 256, 0, stream>>>(Wkv + (size_t)i * D * 2 * D, WkvP + (size_t)i * D * 2 * D, D, 2 * D);
    k_pack_b<<<64, 256, 0, stream>>>(W1  + (size_t)i * D * D,      W1P  + (size_t)i * D * D,     D, D);
    k_pack_b<<<64, 256, 0, stream>>>(W2  + (size_t)i * D * D,      W2P  + (size_t)i * D * D,     D, D);
  }
  k_pack_b<<<1728, 256, 0, stream>>>(convW, convP, 27 * D, D);

  const dim3 gQ(MG / 64, D / 64);       // (2048, 2)
  const dim3 gKV(MG / 64, 2 * D / 64);  // (2048, 4)
  auto layer = [&](const float* QF, const float* KVF, float* OUT, const int* occ, int i) {
    k_gemm128<<<gQ, 256, 0, stream>>>(QF, WqP + (size_t)i * D * D, bq + i * D,
                                      nullptr, q16, D, 0);
    k_gemm128<<<gKV, 256, 0, stream>>>(KVF, WkvP + (size_t)i * D * 2 * D, bkv + i * 2 * D,
                                       nullptr, kv16, 2 * D, 0);
    k_attn<<<dim3(NGRP, H), 128, 0, stream>>>(q16, kv16, occ, featb);
    k_ln_res<<<MG / 8, 256, 0, stream>>>(featb, QF, ln1g + i * D, ln1b + i * D, xln, MG);
    k_gemm128<<<gQ, 256, 0, stream>>>(xln, W1P + (size_t)i * D * D, b1 + i * D,
                                      hbuf, nullptr, D, 1);
    k_gemm128<<<gQ, 256, 0, stream>>>(hbuf, W2P + (size_t)i * D * D, b2 + i * D,
                                      featb, nullptr, D, 0);
    k_ln_res<<<MG / 8, 256, 0, stream>>>(featb, xln, ln2g + i * D, ln2b + i * D, OUT, MG);
  };
  layer(buf0, buf0, buf2, occX, 0);  // X1 = att(Xg, Xg, mX, 0)
  layer(buf1, buf1, buf3, occY, 1);  // Y1 = att(Yg, Yg, mY, 1)
  layer(buf3, buf2, buf0, occX, 2);  // Xo = att(Y1, X1, mX, 2)
  layer(buf2, buf3, buf1, occY, 3);  // Yo = att(X1, Y1, mY, 3)

  // scatter back: xTF = x; xTF[permX] = Xo[slotsX]; xTF[permY] += Yo[slotsY]
  k_copy<<<8192, 256, 0, stream>>>(x, xTF, (long)NPTS * D);
  k_scatter_set<<<8192, 256, 0, stream>>>(permX, slotsX, buf0, xTF, NPTS);
  k_scatter_add<<<8192, 256, 0, stream>>>(permY, slotsY, buf1, xTF, NPTS);

  // f16 pad (row NPTS zeroed) for async gather
  k_f32_to_f16<<<8192, 256, 0, stream>>>(xTF, pad16, (long)NPTS * D);
  k_zero<<<1, 64, 0, stream>>>((float*)(pad16 + (size_t)NPTS * D), 64);

  // neighbor conv + LN + GELU
  k_conv<<<(NPTS + 15) / 16, 256, 0, stream>>>(pad16, nbr, convP, clng, clnb,
                                               (float*)d_out);
}
